// RGBD_RefNet_12498354831815
// MI455X (gfx1250) — matrix operand order repaired and checked by reading, not compiled, not run on hardware
//
#include <hip/hip_runtime.h>
#include <cstdint>
#include <cstddef>

// ============================================================================
// RGBD RefNet forward for MI455X (gfx1250, wave32, WMMA).
//
//  * All MLPs run through k_mlp_chain: per-block 128 rows, 8 waves, each wave
//    computes 16x16 D-tiles via v_wmma_f32_16x16x32_bf16 (bf16 A/B, f32 acc).
//    Activations stay in LDS (bf16, double-buffered); per-n-tile weight slices
//    are cooperatively staged into LDS (transposed, bf16). The WMMA K-loop is
//    software-pipelined (2-stage ping-pong) so ds_load_b128 latency overlaps
//    XDL execution. Optional max-pool epilogue.
//  * FPS: one workgroup/batch, sequential scan, LDS argmax reduction.
//  * Ball query: thread/query, first-k-in-radius scan (matches top_k(-cand)).
//
// Input order assumption (dict insertion order, params pytree in insertion
// order, (W,b) per layer):
//   0: input_point_cloud (B,3,N)  1: lang_feat (B,256)  2: pcd_heatmap (B,N)
//   3..: enc1(6) enc2(6) enc3(6) agg_mlp(4) agg_post(2) fuse(6) vgen(6)
//        vote_sa(6) intact_mlp(6) intact_post(2) res1_blocks(8) res1_final(2)
//        res2_blocks(8) res2_final(2) head_partial(2) head_intact(2)
//        head_conf(2)                                   -> n_in = 79
// Output (floats, concatenated): cx(192) vote_heat(64) vote_xyz(1536)
//   seed_xyz(1536) partial_loc(384) intact_loc(384) conf(64) pvote_xyz(3072)
//   point_ind(512) vote_inds(64) ind_512(1024)  => 8832 elements.
// ============================================================================

namespace {

constexpr int Bb = 2;
constexpr int Np = 20000;

typedef __attribute__((ext_vector_type(16))) __bf16 v16bf;
typedef __attribute__((ext_vector_type(8)))  float  v8f;

struct ChainArgs {
  const float* X;
  float* Y;
  const float* W[3];
  const float* Bv[3];
  int cin[3];
  int cout[3];
  int acts[3];   // 0 none, 1 relu, 2 sigmoid
  int nlayers;
  int Mrows;
  int pool_k;    // 0 = none; else max-pool over groups of pool_k rows
  int cpad;      // LDS activation row stride in halves (pad32(max dim) + 16)
};

// ---------------------------------------------------------------------------
// Fused multi-layer MLP (+optional max-pool) on 128-row blocks. WMMA bf16.
// ---------------------------------------------------------------------------
__global__ void k_mlp_chain(ChainArgs a) {
  extern __shared__ char smem_raw[];
  __bf16* bufA = (__bf16*)smem_raw;
  __bf16* bufB = bufA + 128 * a.cpad;
  __bf16* Wt   = bufB + 128 * a.cpad;   // 16 x (kpad+16) weight tile

  const int tid  = threadIdx.x;         // 256 threads = 8 wave32 waves
  const int wave = tid >> 5;
  const int lane = tid & 31;
  const int h    = lane >> 4;           // lane half (WMMA layout)
  const int ln   = lane & 15;
  const int rowBlock = blockIdx.x * 128;

  // Stage input rows (zero-padded) into LDS as bf16. Branchless clamp+select.
  {
    const int c0  = a.cin[0];
    const int c0p = (c0 + 31) & ~31;
    for (int i = tid; i < 128 * c0p; i += 256) {
      int r = i / c0p, c = i - r * c0p;
      int gr = rowBlock + r;
      int grc = gr < a.Mrows ? gr : a.Mrows - 1;
      int cc  = c < c0 ? c : c0 - 1;
      float v = a.X[(size_t)grc * c0 + cc];
      if (gr >= a.Mrows || c >= c0) v = 0.f;
      bufA[(size_t)r * a.cpad + c] = (__bf16)v;
    }
  }
  __syncthreads();

  __bf16* cur = bufA;
  __bf16* nxt = bufB;

  for (int l = 0; l < a.nlayers; ++l) {
    const int K      = a.cin[l];
    const int Nn     = a.cout[l];
    const int kpad   = (K + 31) & ~31;
    const int wstr   = kpad + 16;       // Wt row stride (halves, 32B-aligned)
    const int ntiles = (Nn + 15) >> 4;
    const int nchunk = kpad >> 5;       // number of 32-wide K chunks
    const bool lastDirect = (l == a.nlayers - 1) && (a.pool_k == 0);
    const float* Wl = a.W[l];
    const float* bl = a.Bv[l];
    const int act   = a.acts[l];
    const int mrow0 = wave * 16;        // this wave's M-tile within the block

    for (int nt = 0; nt < ntiles; ++nt) {
      // ---- cooperative staging of W[:, nt*16 .. +16) transposed to LDS ----
      __syncthreads();                  // previous tile's readers are done
      {
        const int n0 = nt * 16;
        for (int i = tid; i < 16 * kpad; i += 256) {
          int kr = i >> 4, nl_ = i & 15;          // coalesced over columns
          int n  = n0 + nl_;
          int krc = kr < K ? kr : K - 1;
          int nc  = n < Nn ? n : Nn - 1;
          float wv = Wl[(size_t)krc * Nn + nc];
          if (kr >= K || n >= Nn) wv = 0.f;
          Wt[(size_t)nl_ * wstr + kr] = (__bf16)wv;
        }
      }
      __syncthreads();

      // ---- WMMA K-loop: LDS-only, 2-stage software pipeline ----------------
      const __bf16* aBase = cur + (size_t)(mrow0 + ln) * a.cpad + 16 * h;
      const __bf16* bBase = Wt + (size_t)ln * wstr + 16 * h;
      auto loadFrag = [&](int k0, v16bf& af, v16bf& bf) {
        const __bf16* ar = aBase + k0;   // A: halves = K(16h + i)
#pragma unroll
        for (int i = 0; i < 8; ++i) af[i] = ar[i];
#pragma unroll
        for (int i = 0; i < 8; ++i) af[8 + i] = ar[8 + i];
        const __bf16* br = bBase + k0;   // B: halves = K(16h + i)
#pragma unroll
        for (int i = 0; i < 16; ++i) bf[i] = br[i];
      };

      v8f acc = {0.f, 0.f, 0.f, 0.f, 0.f, 0.f, 0.f, 0.f};
      v16bf af0, bf0, af1, bf1;
      loadFrag(0, af0, bf0);
      int c = 0;
      for (; c + 2 <= nchunk; c += 2) {
        loadFrag((c + 1) * 32, af1, bf1);      // prefetch next chunk
        acc = __builtin_amdgcn_wmma_f32_16x16x32_bf16(
            false, af0, false, bf0, (short)0, acc, false, false);
        if (c + 2 < nchunk) loadFrag((c + 2) * 32, af0, bf0);
        acc = __builtin_amdgcn_wmma_f32_16x16x32_bf16(
            false, af1, false, bf1, (short)0, acc, false, false);
      }
      if (c < nchunk) {                        // odd tail chunk (in af0/bf0)
        acc = __builtin_amdgcn_wmma_f32_16x16x32_bf16(
            false, af0, false, bf0, (short)0, acc, false, false);
      }

      // ---- epilogue: bias + activation (C/D: lane=col, VGPR r = row r+8h) --
      const int n = nt * 16 + ln;
      float bias = bl[n < Nn ? n : 0];
      if (n >= Nn) bias = 0.f;
#pragma unroll
      for (int r = 0; r < 8; ++r) {
        int lr = mrow0 + r + 8 * h;
        float v = acc[r] + bias;
        if (act == 1)      v = fmaxf(v, 0.f);
        else if (act == 2) v = __builtin_amdgcn_rcpf(1.f + __expf(-v));
        if (lastDirect) {
          int gr = rowBlock + lr;
          if (gr < a.Mrows && n < Nn) a.Y[(size_t)gr * Nn + n] = v;
        } else if (n < Nn) {
          nxt[(size_t)lr * a.cpad + n] = (__bf16)v;
        }
      }
    }

    // Zero-pad columns [Nn, pad32(Nn)) of the produced activation.
    if (!lastDirect && l + 1 < a.nlayers) {
      int np_ = (Nn + 31) & ~31;
      int padw = np_ - Nn;
      if (padw > 0) {
        for (int i = tid; i < 128 * padw; i += 256) {
          int r = i / padw, c2 = Nn + (i - r * padw);
          nxt[(size_t)r * a.cpad + c2] = (__bf16)0.f;
        }
      }
    }
    __syncthreads();
    __bf16* t = cur; cur = nxt; nxt = t;
  }

  // Max-pool epilogue over groups of pool_k rows (final layer lives in `cur`).
  if (a.pool_k > 0) {
    const int Cout   = a.cout[a.nlayers - 1];
    const int groups = 128 / a.pool_k;
    const int gBase  = rowBlock / a.pool_k;
    const int totG   = a.Mrows / a.pool_k;
    for (int i = tid; i < groups * Cout; i += 256) {
      int g = i / Cout, c = i - g * Cout;
      int gg = gBase + g;
      if (gg < totG) {
        float m = -3.4e38f;
        for (int j = 0; j < a.pool_k; ++j)
          m = fmaxf(m, (float)cur[(size_t)(g * a.pool_k + j) * a.cpad + c]);
        a.Y[(size_t)gg * Cout + c] = m;
      }
    }
  }
}

// ---------------------------------------------------------------------------
// Transpose (B,3,N) -> (B,N,3)
// ---------------------------------------------------------------------------
__global__ void k_transpose(const float* in, float* out, int n) {
  int i = blockIdx.x * blockDim.x + threadIdx.x;
  if (i >= n) return;
  int b = i / Np, j = i - b * Np;
  const float* p = in + (size_t)b * 3 * Np + j;
  float* o = out + (size_t)i * 3;
  o[0] = p[0]; o[1] = p[Np]; o[2] = p[2 * Np];
}

// ---------------------------------------------------------------------------
// Farthest point sampling. One workgroup per batch; sequential over npoint.
// useW: score = dist * clamp(heat, 0.1, 1.0)  (ALPHA=1).
// ---------------------------------------------------------------------------
__global__ void k_fps(const float* xyz, const float* heat, int useW,
                      int* out_idx, float* dist, int N, int npoint) {
  const int b = blockIdx.x;
  const int tid = threadIdx.x;
  const int nt = blockDim.x;               // 512
  __shared__ float sv[512];
  __shared__ int   si[512];
  __shared__ int   s_cur;
  const float* P = xyz + (size_t)b * N * 3;
  float* D = dist + (size_t)b * N;
  const float* W = useW ? (heat + (size_t)b * N) : nullptr;
  for (int j = tid; j < N; j += nt) D[j] = 1e10f;
  if (tid == 0) { out_idx[(size_t)b * npoint] = 0; s_cur = 0; }
  __syncthreads();
  for (int s = 1; s < npoint; ++s) {
    const int cur = s_cur;
    const float px = P[cur * 3], py = P[cur * 3 + 1], pz = P[cur * 3 + 2];
    float bv = -1.f; int bi = 0;
    for (int j = tid; j < N; j += nt) {
      float dx = P[j * 3] - px, dy = P[j * 3 + 1] - py, dz = P[j * 3 + 2] - pz;
      float d = fminf(D[j], dx * dx + dy * dy + dz * dz);
      D[j] = d;
      float sc = d;
      if (useW) sc = d * fminf(fmaxf(W[j], 0.1f), 1.f);
      if (sc > bv) { bv = sc; bi = j; }     // strict > keeps lowest index
    }
    sv[tid] = bv; si[tid] = bi;
    __syncthreads();
    for (int off = nt >> 1; off > 0; off >>= 1) {
      if (tid < off) {
        if (sv[tid + off] > sv[tid] ||
            (sv[tid + off] == sv[tid] && si[tid + off] < si[tid])) {
          sv[tid] = sv[tid + off]; si[tid] = si[tid + off];
        }
      }
      __syncthreads();
    }
    if (tid == 0) { s_cur = si[0]; out_idx[(size_t)b * npoint + s] = si[0]; }
    __syncthreads();
  }
}

// ---------------------------------------------------------------------------
// Ball query: first k in-radius indices (ascending) == top_k(-cand) semantics.
// ---------------------------------------------------------------------------
__global__ void k_ball(const float* q, const float* s, int M, int N,
                       float r2, int k, int* out, int total) {
  int i = blockIdx.x * blockDim.x + threadIdx.x;
  if (i >= total) return;                   // total = Bb*M
  int b = i / M;
  const float* Q = q + (size_t)i * 3;
  const float* S = s + (size_t)b * N * 3;
  int* O = out + (size_t)i * k;
  const float qx = Q[0], qy = Q[1], qz = Q[2];
  int cnt = 0;
  for (int j = 0; j < N && cnt < k; ++j) {
    float dx = S[j * 3] - qx, dy = S[j * 3 + 1] - qy, dz = S[j * 3 + 2] - qz;
    if (dx * dx + dy * dy + dz * dz <= r2) O[cnt++] = j;
  }
  if (cnt == 0) { for (int t = 0; t < k; ++t) O[t] = N - 1; }
  else { int f = O[0]; for (int t = cnt; t < k; ++t) O[t] = f; }
}

// ---------------------------------------------------------------------------
// Build grouped features: optional (xyz - center)*invR prefix, then gathered
// feature channels. out row = ((b*M + m)*k + j), stride Cin.
// ---------------------------------------------------------------------------
__global__ void k_group(const float* sxyz, const float* feats,
                        const float* centers, const int* idx,
                        int M, int N, int k, int Cf, int useXyz, float invR,
                        float* out, int Cin, int total) {
  int i = blockIdx.x * blockDim.x + threadIdx.x;
  if (i >= total) return;                   // total = Bb*M*k
  int bm = i / k;
  int b = bm / M;
  int src = idx[i];
  float* O = out + (size_t)i * Cin;
  int c0 = 0;
  if (useXyz) {
    const float* S = sxyz + ((size_t)b * N + src) * 3;
    const float* C = centers + (size_t)bm * 3;
    O[0] = (S[0] - C[0]) * invR;
    O[1] = (S[1] - C[1]) * invR;
    O[2] = (S[2] - C[2]) * invR;
    c0 = 3;
  }
  if (Cf > 0) {
    const float* F = feats + ((size_t)b * N + src) * Cf;
    for (int c = 0; c < Cf; ++c) O[c0 + c] = F[c];
  }
}

__global__ void k_gather_rows(const float* src, const int* idx, int M, int Nsrc,
                              int C, float* out, int totalElems) {
  int i = blockIdx.x * blockDim.x + threadIdx.x;
  if (i >= totalElems) return;              // Bb*M*C
  int bm = i / C, c = i - bm * C;
  int b = bm / M;
  out[i] = src[((size_t)b * Nsrc + idx[bm]) * C + c];
}

__global__ void k_gather_idx(const int* src, const int* idx, int M, int Nsrc,
                             int* out, int total) {
  int i = blockIdx.x * blockDim.x + threadIdx.x;
  if (i >= total) return;                   // Bb*M
  int b = i / M;
  out[i] = src[(size_t)b * Nsrc + idx[i]];
}

__global__ void k_concat2(const float* A, int C1, const float* Bm, int C2,
                          float* out, int rows) {
  int C = C1 + C2;
  int i = blockIdx.x * blockDim.x + threadIdx.x;
  if (i >= rows * C) return;
  int r = i / C, c = i - r * C;
  out[i] = (c < C1) ? A[(size_t)r * C1 + c] : Bm[(size_t)r * C2 + (c - C1)];
}

__global__ void k_concat_bcast(const float* A, int C1, const float* lang,
                               int C2, int Mper, float* out, int rows) {
  int C = C1 + C2;
  int i = blockIdx.x * blockDim.x + threadIdx.x;
  if (i >= rows * C) return;
  int r = i / C, c = i - r * C;
  int b = r / Mper;
  out[i] = (c < C1) ? A[(size_t)r * C1 + c] : lang[(size_t)b * C2 + (c - C1)];
}

__global__ void k_add_slice3(const float* base3, const float* big,
                             int strideBig, float* out, int rows) {
  int i = blockIdx.x * blockDim.x + threadIdx.x;
  if (i >= rows * 3) return;
  int r = i / 3, c = i - r * 3;
  out[i] = base3[i] + big[(size_t)r * strideBig + c];
}

// vfeat = normalize(fuse + net[:,3:515]) per row (512 channels).
__global__ void k_vote_feat(const float* fuse, const float* net, float* out,
                            int rows) {
  int r = blockIdx.x * blockDim.x + threadIdx.x;
  if (r >= rows) return;
  const float* F = fuse + (size_t)r * 512;
  const float* Nt = net + (size_t)r * 515 + 3;
  float ss = 0.f;
  for (int c = 0; c < 512; ++c) { float v = F[c] + Nt[c]; ss += v * v; }
  float inv = rsqrtf(ss);
  float* O = out + (size_t)r * 512;
  for (int c = 0; c < 512; ++c) O[c] = (F[c] + Nt[c]) * inv;
}

__global__ void k_add_relu(float* x, const float* y, int n) {
  int i = blockIdx.x * blockDim.x + threadIdx.x;
  if (i < n) x[i] = fmaxf(x[i] + y[i], 0.f);
}

__global__ void k_copy_f(const float* s, float* d, int n) {
  int i = blockIdx.x * blockDim.x + threadIdx.x;
  if (i < n) d[i] = s[i];
}

__global__ void k_i2f(const int* s, float* d, int n) {
  int i = blockIdx.x * blockDim.x + threadIdx.x;
  if (i < n) d[i] = (float)s[i];
}

inline int ceilDiv(int a, int b) { return (a + b - 1) / b; }

void run_chain(hipStream_t st, const float* X, float* Y, int Mrows, int pool_k,
               int nl,
               const float* W0, const float* b0, int i0, int o0, int a0,
               const float* W1 = nullptr, const float* b1 = nullptr,
               int i1 = 0, int o1 = 0, int a1 = 0,
               const float* W2 = nullptr, const float* b2 = nullptr,
               int i2 = 0, int o2 = 0, int a2 = 0) {
  ChainArgs a{};
  a.X = X; a.Y = Y; a.Mrows = Mrows; a.pool_k = pool_k; a.nlayers = nl;
  a.W[0] = W0; a.Bv[0] = b0; a.cin[0] = i0; a.cout[0] = o0; a.acts[0] = a0;
  a.W[1] = W1; a.Bv[1] = b1; a.cin[1] = i1; a.cout[1] = o1; a.acts[1] = a1;
  a.W[2] = W2; a.Bv[2] = b2; a.cin[2] = i2; a.cout[2] = o2; a.acts[2] = a2;
  int maxpad = 32, kpadmax = 32;
  for (int l = 0; l < nl; ++l) {
    int p1 = (a.cin[l] + 31) & ~31, p2 = (a.cout[l] + 31) & ~31;
    if (p1 > maxpad) maxpad = p1;
    if (p2 > maxpad) maxpad = p2;
    if (p1 > kpadmax) kpadmax = p1;
  }
  a.cpad = maxpad + 16;                       // +16 halves: bank-conflict pad
  size_t smem = ((size_t)2 * 128 * a.cpad + (size_t)16 * (kpadmax + 16)) * 2;
  k_mlp_chain<<<dim3(ceilDiv(Mrows, 128)), dim3(256), smem, st>>>(a);
}

}  // namespace

// ============================================================================
extern "C" void kernel_launch(void* const* d_in, const int* in_sizes, int n_in,
                              void* d_out, int out_size, void* d_ws,
                              size_t ws_size, hipStream_t stream) {
  const float* pc   = (const float*)d_in[0];  // (B,3,N)
  const float* lang = (const float*)d_in[1];  // (B,256)
  const float* heat = (const float*)d_in[2];  // (B,N)
  auto PP = [&](int i) { return (const float*)d_in[3 + i]; };

  // Param leaf indices (relative to d_in[3]).
  const float *e1W0=PP(0),*e1b0=PP(1),*e1W1=PP(2),*e1b1=PP(3),*e1W2=PP(4),*e1b2=PP(5);
  const float *e2W0=PP(6),*e2b0=PP(7),*e2W1=PP(8),*e2b1=PP(9),*e2W2=PP(10),*e2b2=PP(11);
  const float *e3W0=PP(12),*e3b0=PP(13),*e3W1=PP(14),*e3b1=PP(15),*e3W2=PP(16),*e3b2=PP(17);
  const float *agW0=PP(18),*agb0=PP(19),*agW1=PP(20),*agb1=PP(21);
  const float *apW=PP(22),*apb=PP(23);
  const float *fuW0=PP(24),*fub0=PP(25),*fuW1=PP(26),*fub1=PP(27),*fuW2=PP(28),*fub2=PP(29);
  const float *vgW0=PP(30),*vgb0=PP(31),*vgW1=PP(32),*vgb1=PP(33),*vgW2=PP(34),*vgb2=PP(35);
  const float *vsW0=PP(36),*vsb0=PP(37),*vsW1=PP(38),*vsb1=PP(39),*vsW2=PP(40),*vsb2=PP(41);
  const float *imW0=PP(42),*imb0=PP(43),*imW1=PP(44),*imb1=PP(45),*imW2=PP(46),*imb2=PP(47);
  const float *ipW=PP(48),*ipb=PP(49);
  const float *r1a1W=PP(50),*r1a1b=PP(51),*r1a2W=PP(52),*r1a2b=PP(53);
  const float *r1b1W=PP(54),*r1b1b=PP(55),*r1b2W=PP(56),*r1b2b=PP(57);
  const float *r1fW=PP(58),*r1fb=PP(59);
  const float *r2a1W=PP(60),*r2a1b=PP(61),*r2a2W=PP(62),*r2a2b=PP(63);
  const float *r2b1W=PP(64),*r2b1b=PP(65),*r2b2W=PP(66),*r2b2b=PP(67);
  const float *r2fW=PP(68),*r2fb=PP(69);
  const float *hpW=PP(70),*hpb=PP(71);
  const float *hiW=PP(72),*hib=PP(73);
  const float *hcW=PP(74),*hcb=PP(75);

  // Workspace bump allocator.
  char* wsB = (char*)d_ws;
  size_t off = 0;
  auto allocB = [&](size_t bytes) -> void* {
    void* p = wsB + off;
    off = (off + bytes + 255) & ~(size_t)255;
    return p;
  };
  auto aF = [&](size_t n) { return (float*)allocB(n * 4); };
  auto aI = [&](size_t n) { return (int*)allocB(n * 4); };

  float* xyz   = aF((size_t)Bb * Np * 3);
  float* dist  = aF((size_t)Bb * Np);
  int*   fidx1 = aI(Bb * 2048);
  float* x1    = aF(Bb * 2048 * 3);
  int*   idx1  = aI((size_t)Bb * 2048 * 64);
  float* g1    = aF((size_t)Bb * 2048 * 64 * 3);
  float* f1    = aF((size_t)Bb * 2048 * 128);
  int*   fidx2 = aI(Bb * 1024);
  float* x2    = aF(Bb * 1024 * 3);
  int*   idx2  = aI((size_t)Bb * 1024 * 32);
  float* g2    = aF((size_t)Bb * 1024 * 32 * 128);
  float* f2    = aF((size_t)Bb * 1024 * 256);
  int*   fidx3 = aI(Bb * 512);
  float* keyx  = aF(Bb * 512 * 3);
  int*   idx3  = aI((size_t)Bb * 512 * 16);
  float* g3    = aF((size_t)Bb * 512 * 16 * 256);
  float* pw    = aF((size_t)Bb * 512 * 256);
  int*   tmpI  = aI(Bb * 512);
  int*   ind512= aI(Bb * 512);
  int*   pind  = aI(Bb * 256);
  float* seed  = aF(Bb * 256 * 3);
  int*   idxs  = aI((size_t)Bb * 256 * 16);
  float* gs    = aF((size_t)Bb * 256 * 16 * 256);
  float* hs    = aF((size_t)Bb * 256 * 256);
  float* cats  = aF((size_t)Bb * 256 * 259);
  float* sfeat = aF((size_t)Bb * 256 * 256);
  float* fusein= aF((size_t)Bb * 256 * 512);
  float* fuseo = aF((size_t)Bb * 256 * 512);
  float* net   = aF((size_t)Bb * 256 * 515);
  float* vxyz  = aF(Bb * 256 * 3);
  float* vfeat = aF((size_t)Bb * 256 * 512);
  int*   fidxv = aI(Bb * 32);
  float* cx    = aF(Bb * 32 * 3);
  int*   idxv  = aI(Bb * 32 * 16);
  float* gv    = aF((size_t)Bb * 32 * 16 * 515);
  float* cf    = aF(Bb * 32 * 128);
  float* yres  = aF(Bb * 32 * 128);
  float* featb = aF(Bb * 32 * 128);
  float* ploc  = aF(Bb * 32 * 6);
  float* conf  = aF(Bb * 32);
  float* pcen  = aF(Bb * 32 * 3);
  int*   idxi  = aI(Bb * 32 * 32);
  float* gi    = aF((size_t)Bb * 32 * 32 * 512);
  float* hi    = aF(Bb * 32 * 128);
  float* cati  = aF(Bb * 32 * 256);
  float* ifeat = aF(Bb * 32 * 128);
  float* ifeatF= aF(Bb * 32 * 128);
  float* iloc  = aF(Bb * 32 * 6);
  float* vheat = aF(Bb * 32);
  int*   vinds = aI(Bb * 32);
  int*   idxp  = aI((size_t)Bb * 512 * 16);
  float* gp    = aF((size_t)Bb * 512 * 16 * 256);
  float* hp    = aF((size_t)Bb * 512 * 256);
  float* catp  = aF((size_t)Bb * 512 * 259);
  float* psf   = aF((size_t)Bb * 512 * 256);
  float* pfin  = aF((size_t)Bb * 512 * 512);
  float* pfuse = aF((size_t)Bb * 512 * 512);
  float* pnet  = aF((size_t)Bb * 512 * 515);
  float* pvote = aF(Bb * 512 * 3);
  (void)ws_size; (void)in_sizes; (void)n_in; (void)out_size;

#define EL(kern, n, ...)                                                    \
  do { int _n = (n);                                                        \
       if (_n > 0) kern<<<dim3((_n + 255) / 256), dim3(256), 0, stream>>>(  \
           __VA_ARGS__); } while (0)

  // ---- xyz transpose -------------------------------------------------------
  EL(k_transpose, Bb * Np, pc, xyz, Bb * Np);

  // ---- enc1: SA(2048, r=0.1, k=64, [3,64,64,128], use_xyz) ----------------
  k_fps<<<dim3(Bb), dim3(512), 0, stream>>>(xyz, nullptr, 0, fidx1, dist, Np, 2048);
  EL(k_gather_rows, Bb * 2048 * 3, xyz, fidx1, 2048, Np, 3, x1, Bb * 2048 * 3);
  EL(k_ball, Bb * 2048, x1, xyz, 2048, Np, 0.01f, 64, idx1, Bb * 2048);
  EL(k_group, Bb * 2048 * 64, xyz, nullptr, x1, idx1, 2048, Np, 64, 0, 1, 1.f,
     g1, 3, Bb * 2048 * 64);
  run_chain(stream, g1, f1, Bb * 2048 * 64, 64, 3,
            e1W0, e1b0, 3, 64, 1,  e1W1, e1b1, 64, 64, 1,  e1W2, e1b2, 64, 128, 1);

  // ---- enc2: SA(1024, r=0.2, k=32, [128,128,128,256]) ---------------------
  k_fps<<<dim3(Bb), dim3(512), 0, stream>>>(x1, nullptr, 0, fidx2, dist, 2048, 1024);
  EL(k_gather_rows, Bb * 1024 * 3, x1, fidx2, 1024, 2048, 3, x2, Bb * 1024 * 3);
  EL(k_ball, Bb * 1024, x2, x1, 1024, 2048, 0.04f, 32, idx2, Bb * 1024);
  EL(k_group, Bb * 1024 * 32, x1, f1, nullptr, idx2, 1024, 2048, 32, 128, 0, 1.f,
     g2, 128, Bb * 1024 * 32);
  run_chain(stream, g2, f2, Bb * 1024 * 32, 32, 3,
            e2W0, e2b0, 128, 128, 1, e2W1, e2b1, 128, 128, 1, e2W2, e2b2, 128, 256, 1);

  // ---- enc3: SA(512, r=0.4, k=16, [256,128,128,256]) ----------------------
  k_fps<<<dim3(Bb), dim3(512), 0, stream>>>(x2, nullptr, 0, fidx3, dist, 1024, 512);
  EL(k_gather_rows, Bb * 512 * 3, x2, fidx3, 512, 1024, 3, keyx, Bb * 512 * 3);
  EL(k_ball, Bb * 512, keyx, x2, 512, 1024, 0.16f, 16, idx3, Bb * 512);
  EL(k_group, Bb * 512 * 16, x2, f2, nullptr, idx3, 512, 1024, 16, 256, 0, 1.f,
     g3, 256, Bb * 512 * 16);
  run_chain(stream, g3, pw, Bb * 512 * 16, 16, 3,
            e3W0, e3b0, 256, 128, 1, e3W1, e3b1, 128, 128, 1, e3W2, e3b2, 128, 256, 1);

  // ind_512 = i1[i2[i3]]
  EL(k_gather_idx, Bb * 512, fidx2, fidx3, 512, 1024, tmpI, Bb * 512);
  EL(k_gather_idx, Bb * 512, fidx1, tmpI, 512, 2048, ind512, Bb * 512);

  // ---- weighted FPS seeds + seed aggregation ------------------------------
  k_fps<<<dim3(Bb), dim3(512), 0, stream>>>(xyz, heat, 1, pind, dist, Np, 256);
  EL(k_gather_rows, Bb * 256 * 3, xyz, pind, 256, Np, 3, seed, Bb * 256 * 3);
  EL(k_ball, Bb * 256, seed, keyx, 256, 512, 0.16f, 16, idxs, Bb * 256);
  EL(k_group, Bb * 256 * 16, keyx, pw, nullptr, idxs, 256, 512, 16, 256, 0, 1.f,
     gs, 256, Bb * 256 * 16);
  run_chain(stream, gs, hs, Bb * 256 * 16, 16, 2,
            agW0, agb0, 256, 256, 1, agW1, agb1, 256, 256, 1);
  EL(k_concat2, Bb * 256 * 259, hs, 256, seed, 3, cats, Bb * 256);
  run_chain(stream, cats, sfeat, Bb * 256, 0, 1, apW, apb, 259, 256, 1);

  // ---- language fusion + voting -------------------------------------------
  EL(k_concat_bcast, Bb * 256 * 512, sfeat, 256, lang, 256, 256, fusein, Bb * 256);
  run_chain(stream, fusein, fuseo, Bb * 256, 0, 3,
            fuW0, fub0, 512, 512, 1, fuW1, fub1, 512, 512, 1, fuW2, fub2, 512, 512, 0);
  run_chain(stream, fuseo, net, Bb * 256, 0, 3,
            vgW0, vgb0, 512, 512, 1, vgW1, vgb1, 512, 512, 1, vgW2, vgb2, 512, 515, 0);
  EL(k_add_slice3, Bb * 256 * 3, seed, net, 515, vxyz, Bb * 256);
  EL(k_vote_feat, Bb * 256, fuseo, net, vfeat, Bb * 256);

  // ---- vote SA(32, r=0.3, k=16, [515,256,128,128], use_xyz, norm) ---------
  k_fps<<<dim3(Bb), dim3(512), 0, stream>>>(vxyz, nullptr, 0, fidxv, dist, 256, 32);
  EL(k_gather_rows, Bb * 32 * 3, vxyz, fidxv, 32, 256, 3, cx, Bb * 32 * 3);
  EL(k_ball, Bb * 32, cx, vxyz, 32, 256, 0.09f, 16, idxv, Bb * 32);
  EL(k_group, Bb * 32 * 16, vxyz, vfeat, cx, idxv, 32, 256, 16, 512, 1,
     1.f / 0.3f, gv, 515, Bb * 32 * 16);
  run_chain(stream, gv, cf, Bb * 32 * 16, 16, 3,
            vsW0, vsb0, 515, 256, 1, vsW1, vsb1, 256, 128, 1, vsW2, vsb2, 128, 128, 1);
  EL(k_gather_idx, Bb * 32, pind, fidxv, 32, 256, vinds, Bb * 32);

  // ---- res1 + heads --------------------------------------------------------
  run_chain(stream, cf, yres, Bb * 32, 0, 2,
            r1a1W, r1a1b, 128, 128, 1, r1a2W, r1a2b, 128, 128, 0);
  EL(k_add_relu, Bb * 32 * 128, cf, yres, Bb * 32 * 128);
  run_chain(stream, cf, yres, Bb * 32, 0, 2,
            r1b1W, r1b1b, 128, 128, 1, r1b2W, r1b2b, 128, 128, 0);
  EL(k_add_relu, Bb * 32 * 128, cf, yres, Bb * 32 * 128);
  run_chain(stream, cf, featb, Bb * 32, 0, 1, r1fW, r1fb, 128, 128, 1);
  run_chain(stream, featb, ploc, Bb * 32, 0, 1, hpW, hpb, 128, 6, 0);
  run_chain(stream, featb, conf, Bb * 32, 0, 1, hcW, hcb, 128, 1, 2);

  // ---- intact branch -------------------------------------------------------
  EL(k_add_slice3, Bb * 32 * 3, cx, ploc, 6, pcen, Bb * 32);
  EL(k_ball, Bb * 32, pcen, seed, 32, 256, 0.16f, 32, idxi, Bb * 32);
  EL(k_group, Bb * 32 * 32, seed, fuseo, nullptr, idxi, 32, 256, 32, 512, 0, 1.f,
     gi, 512, Bb * 32 * 32);
  run_chain(stream, gi, hi, Bb * 32 * 32, 32, 3,
            imW0, imb0, 512, 512, 1, imW1, imb1, 512, 256, 1, imW2, imb2, 256, 128, 1);
  EL(k_concat2, Bb * 32 * 256, hi, 128, featb, 128, cati, Bb * 32);
  run_chain(stream, cati, ifeat, Bb * 32, 0, 1, ipW, ipb, 256, 128, 1);
  run_chain(stream, ifeat, yres, Bb * 32, 0, 2,
            r2a1W, r2a1b, 128, 128, 1, r2a2W, r2a2b, 128, 128, 0);
  EL(k_add_relu, Bb * 32 * 128, ifeat, yres, Bb * 32 * 128);
  run_chain(stream, ifeat, yres, Bb * 32, 0, 2,
            r2b1W, r2b1b, 128, 128, 1, r2b2W, r2b2b, 128, 128, 0);
  EL(k_add_relu, Bb * 32 * 128, ifeat, yres, Bb * 32 * 128);
  run_chain(stream, ifeat, ifeatF, Bb * 32, 0, 1, r2fW, r2fb, 128, 128, 1);
  run_chain(stream, ifeatF, iloc, Bb * 32, 0, 1, hiW, hib, 128, 6, 0);

  // ---- vote_heat -----------------------------------------------------------
  EL(k_gather_rows, Bb * 32, heat, vinds, 32, Np, 1, vheat, Bb * 32);

  // ---- pseed / pvote branch ------------------------------------------------
  EL(k_ball, Bb * 512, keyx, keyx, 512, 512, 0.16f, 16, idxp, Bb * 512);
  EL(k_group, Bb * 512 * 16, keyx, pw, nullptr, idxp, 512, 512, 16, 256, 0, 1.f,
     gp, 256, Bb * 512 * 16);
  run_chain(stream, gp, hp, Bb * 512 * 16, 16, 2,
            agW0, agb0, 256, 256, 1, agW1, agb1, 256, 256, 1);
  EL(k_concat2, Bb * 512 * 259, hp, 256, keyx, 3, catp, Bb * 512);
  run_chain(stream, catp, psf, Bb * 512, 0, 1, apW, apb, 259, 256, 1);
  EL(k_concat_bcast, Bb * 512 * 512, psf, 256, lang, 256, 512, pfin, Bb * 512);
  run_chain(stream, pfin, pfuse, Bb * 512, 0, 3,
            fuW0, fub0, 512, 512, 1, fuW1, fub1, 512, 512, 1, fuW2, fub2, 512, 512, 0);
  run_chain(stream, pfuse, pnet, Bb * 512, 0, 3,
            vgW0, vgb0, 512, 512, 1, vgW1, vgb1, 512, 512, 1, vgW2, vgb2, 512, 515, 0);
  EL(k_add_slice3, Bb * 512 * 3, keyx, pnet, 515, pvote, Bb * 512);

  // ---- assemble outputs ----------------------------------------------------
  float* out = (float*)d_out;
  EL(k_copy_f, 192,  cx,    out + 0,    192);
  EL(k_copy_f, 64,   vheat, out + 192,  64);
  EL(k_copy_f, 1536, vxyz,  out + 256,  1536);
  EL(k_copy_f, 1536, seed,  out + 1792, 1536);
  EL(k_copy_f, 384,  ploc,  out + 3328, 384);
  EL(k_copy_f, 384,  iloc,  out + 3712, 384);
  EL(k_copy_f, 64,   conf,  out + 4096, 64);
  EL(k_copy_f, 3072, pvote, out + 4160, 3072);
  EL(k_i2f,    512,  pind,  out + 7232, 512);
  EL(k_i2f,    64,   vinds, out + 7744, 64);
  EL(k_i2f,    1024, ind512,out + 7808, 1024);
#undef EL
}
